// GraphAttentionLayer_13881334301310
// MI455X (gfx1250) — compile-verified, matrix operand order checked by reading, not run on hardware
//
#include <hip/hip_runtime.h>
#include <hip/hip_bf16.h>
#include <cstdint>

typedef _Float16 v16h __attribute__((ext_vector_type(16)));
typedef float    v8f  __attribute__((ext_vector_type(8)));

#define GAT_ALPHA 0.2f
#define GAT_NEG_INF (-9e15f)
#define BCn 64
#define Nn  2048
#define IOn 64

#if __has_builtin(__builtin_amdgcn_s_wait_asynccnt)
#define WAIT_ASYNC(n) __builtin_amdgcn_s_wait_asynccnt(n)
#else
#define WAIT_ASYNC(n) asm volatile("s_wait_asynccnt " #n ::: "memory")
#endif

// ---------------- Kernel 1: h_mean[n,i] = mean over 64 (b,c) slices ----------------
__global__ void k_hmean(const float* __restrict__ h, float* __restrict__ hm) {
    int idx = blockIdx.x * 256 + threadIdx.x;            // 0 .. N*IO-1 (131072)
    float s = 0.f;
#pragma unroll 8
    for (int bc = 0; bc < BCn; ++bc) s += h[(size_t)bc * (Nn * IOn) + idx];
    hm[idx] = s * (1.0f / 64.0f);
}

// ---------------- Kernel 2: s1[n] = h_mean[n]·(W a1), s2[n] = h_mean[n]·(W a2) -----
__global__ void k_logits(const float* __restrict__ hm, const float* __restrict__ W,
                         const float* __restrict__ a,
                         float* __restrict__ s1, float* __restrict__ s2) {
    __shared__ float wa1[IOn];
    __shared__ float wa2[IOn];
    int t = threadIdx.x;
    if (t < 64) {
        float s = 0.f;
        for (int o = 0; o < IOn; ++o) s += W[t * IOn + o] * a[o];
        wa1[t] = s;
    } else if (t < 128) {
        int i = t - 64;
        float s = 0.f;
        for (int o = 0; o < IOn; ++o) s += W[i * IOn + o] * a[IOn + o];
        wa2[i] = s;
    }
    __syncthreads();
    int n = blockIdx.x * 256 + t;
    float v1 = 0.f, v2 = 0.f;
    for (int i = 0; i < IOn; ++i) {
        float x = hm[n * IOn + i];
        v1 += x * wa1[i];
        v2 += x * wa2[i];
    }
    s1[n] = v1;
    s2[n] = v2;
}

// ---------------- Kernel 3: masked leaky-relu + row softmax -> P (f16) -------------
__global__ void k_softmax(const float* __restrict__ s1, const float* __restrict__ s2,
                          const int* __restrict__ adj, _Float16* __restrict__ P) {
    __shared__ float red[256];
    int j = blockIdx.x, t = threadIdx.x;
    float sj = s1[j];
    float tl[8];
    float mx = -3.4e38f;
#pragma unroll
    for (int q = 0; q < 8; ++q) {
        int n = q * 256 + t;
        float x = sj + s2[n];
        x = (x >= 0.f) ? x : GAT_ALPHA * x;
        float v = (adj[(size_t)j * Nn + n] > 0) ? x : GAT_NEG_INF;
        tl[q] = v;
        mx = fmaxf(mx, v);
    }
    red[t] = mx; __syncthreads();
    for (int st = 128; st > 0; st >>= 1) {
        if (t < st) red[t] = fmaxf(red[t], red[t + st]);
        __syncthreads();
    }
    mx = red[0]; __syncthreads();
    float sum = 0.f;
#pragma unroll
    for (int q = 0; q < 8; ++q) { tl[q] = __expf(tl[q] - mx); sum += tl[q]; }
    red[t] = sum; __syncthreads();
    for (int st = 128; st > 0; st >>= 1) {
        if (t < st) red[t] += red[t + st];
        __syncthreads();
    }
    float inv = 1.0f / red[0];
#pragma unroll
    for (int q = 0; q < 8; ++q)
        P[(size_t)j * Nn + q * 256 + t] = (_Float16)(tl[q] * inv);
}

// ---------------- Kernel 4: Wh = h@W per bc, stored transposed f16: WhT[bc][o][n] --
__global__ void k_wht(const float* __restrict__ h, const float* __restrict__ W,
                      _Float16* __restrict__ WhT) {
    __shared__ float lh[64 * 65];   // padded: bank = (n+i)%64 -> conflict free
    __shared__ float lw[64 * 64];
    int bc = blockIdx.y;
    int nbase = blockIdx.x * 64;
    int t = threadIdx.x;
    for (int idx = t; idx < 4096; idx += 256) {
        int r = idx >> 6, c = idx & 63;
        lh[r * 65 + c] = h[((size_t)bc * Nn + nbase + r) * IOn + c];
        lw[idx] = W[idx];
    }
    __syncthreads();
    int n = t & 63, og = t >> 6;
    for (int oo = 0; oo < 16; ++oo) {
        int o = og * 16 + oo;
        float acc = 0.f;
#pragma unroll
        for (int i = 0; i < 64; ++i) acc += lh[n * 65 + i] * lw[i * 64 + o];
        WhT[((size_t)bc * IOn + o) * Nn + nbase + n] = (_Float16)acc;
    }
}

// ---------------- Kernel 5: h_prime[bc] = P @ Wh[bc], fused ELU, via WMMA ----------
// grid: (8 j-tiles of 256 rows, 64 bc), 256 threads = 8 waves.
// Each wave: two 16-row j strips x 4 o-tiles (full O=64); B-fragments shared
// across strips. Tiles staged with GLOBAL_LOAD_ASYNC_TO_LDS_B128 into a
// double-buffered LDS, synchronized with s_wait_asynccnt (ASYNCcnt, in-order).
// LDS rows padded to 80B: 16B-aligned b128 targets, conflict-light ds reads.
__global__ void __launch_bounds__(256) k_gemm(const _Float16* __restrict__ P,
                                              const _Float16* __restrict__ WhT,
                                              float* __restrict__ out) {
    // per buffer: P tile 256 rows x 80B = 20480B, WhT tile 64 rows x 80B = 5120B
    __shared__ __align__(16) unsigned char smem[2 * 25600];
    const unsigned smemBase = (unsigned)(size_t)(&smem[0]);

    int t = threadIdx.x;
    int lane = t & 31, wave = t >> 5;
    int bc = blockIdx.y;
    int jbase = blockIdx.x * 256;

    int chunk = t & 3;           // which 16B (8-half) chunk of a 64B row
    int rbase = t >> 2;          // 0..63
    const _Float16* Prow = P + (size_t)(jbase + rbase) * Nn + chunk * 8;
    const _Float16* Wrow = WhT + ((size_t)bc * IOn + rbase) * Nn + chunk * 8;

    auto stage = [&](int buf, int k0) {
        unsigned pB = smemBase + buf * 25600 + rbase * 80 + chunk * 16;
        const _Float16* g = Prow + k0;
#pragma unroll
        for (int pass = 0; pass < 4; ++pass) {   // 4 x 64 = 256 P rows
            asm volatile("global_load_async_to_lds_b128 %0, %1, off"
                         :: "v"(pB + pass * (64 * 80)),
                            "v"(g + (size_t)pass * 64 * Nn)
                         : "memory");
        }
        unsigned wB = smemBase + buf * 25600 + 20480 + rbase * 80 + chunk * 16;
        asm volatile("global_load_async_to_lds_b128 %0, %1, off"
                     :: "v"(wB), "v"(Wrow + k0) : "memory");
    };

    v8f acc[2][4];
#pragma unroll
    for (int s = 0; s < 2; ++s)
#pragma unroll
        for (int ot = 0; ot < 4; ++ot)
#pragma unroll
            for (int r = 0; r < 8; ++r) acc[s][ot][r] = 0.f;

    int uo = (lane >> 4) * 4;    // ISA 16-bit frag: lanes 0-15 K{0-7,16-23}, 16-31 K{8-15,24-31}
    union F { unsigned int u[8]; v16h v; };

    stage(0, 0);
    for (int k0 = 0; k0 < Nn; k0 += 32) {
        int cur = (k0 >> 5) & 1;
        if (k0 + 32 < Nn) {
            stage(cur ^ 1, k0 + 32);   // prefetch next stage (outstanding -> 10)
            WAIT_ASYNC(5);             // oldest 5 (current stage) landed
        } else {
            WAIT_ASYNC(0);
        }
        __syncthreads();

        const unsigned int* lpU = (const unsigned int*)(smem + cur * 25600);
        const unsigned int* lwU = (const unsigned int*)(smem + cur * 25600 + 20480);

        F Bf[4];
#pragma unroll
        for (int ot = 0; ot < 4; ++ot) {
            int orow = ot * 16 + (lane & 15);
#pragma unroll
            for (int v = 0; v < 4; ++v) {
                Bf[ot].u[v]     = lwU[orow * 20 + uo + v];
                Bf[ot].u[4 + v] = lwU[orow * 20 + 8 + uo + v];
            }
        }
#pragma unroll
        for (int s = 0; s < 2; ++s) {
            int arow = wave * 32 + s * 16 + (lane & 15);
            F Af;
#pragma unroll
            for (int v = 0; v < 4; ++v) {
                Af.u[v]     = lpU[arow * 20 + uo + v];
                Af.u[4 + v] = lpU[arow * 20 + 8 + uo + v];
            }
#pragma unroll
            for (int ot = 0; ot < 4; ++ot) {
                acc[s][ot] = __builtin_amdgcn_wmma_f32_16x16x32_f16(
                    false, Af.v, false, Bf[ot].v, (short)0, acc[s][ot], false, false);
            }
        }
        __syncthreads();   // all waves done reading before buffer is re-staged
    }

    // Epilogue: ELU, scatter per C/D layout (VGPR r: lanes0-15 M=r, lanes16-31 M=8+r)
    int mlo = (lane >> 4) * 8;
    int col = lane & 15;
#pragma unroll
    for (int s = 0; s < 2; ++s) {
#pragma unroll
        for (int ot = 0; ot < 4; ++ot) {
#pragma unroll
            for (int r = 0; r < 8; ++r) {
                float x = acc[s][ot][r];
                x = (x > 0.f) ? x : expm1f(x);
                int j = jbase + wave * 32 + s * 16 + mlo + r;
                out[((size_t)bc * Nn + j) * IOn + ot * 16 + col] = x;
            }
        }
    }
}

extern "C" void kernel_launch(void* const* d_in, const int* in_sizes, int n_in,
                              void* d_out, int out_size, void* d_ws, size_t ws_size,
                              hipStream_t stream) {
    const float* h   = (const float*)d_in[0];   // [8,8,2048,64]
    const float* W   = (const float*)d_in[1];   // [64,64]
    const float* a   = (const float*)d_in[2];   // [128,1]
    const int*   adj = (const int*)d_in[3];     // [2048,2048]
    float* out = (float*)d_out;                 // [8,8,2048,64]

    char* ws = (char*)d_ws;
    // workspace layout (offsets 16B-aligned): total ~24.5 MB
    float*    h_mean = (float*)(ws);                               // 524288 B
    float*    s1     = (float*)(ws + 524288);                      //   8192 B
    float*    s2     = (float*)(ws + 532480);                      //   8192 B
    _Float16* P      = (_Float16*)(ws + 540672);                   // 8388608 B
    _Float16* WhT    = (_Float16*)(ws + 540672 + 8388608);         // 16777216 B

    k_hmean  <<<512,           256, 0, stream>>>(h, h_mean);
    k_logits <<<8,             256, 0, stream>>>(h_mean, W, a, s1, s2);
    k_softmax<<<2048,          256, 0, stream>>>(s1, s2, adj, P);
    k_wht    <<<dim3(32, 64),  256, 0, stream>>>(h, W, WhT);
    k_gemm   <<<dim3(8, 64),   256, 0, stream>>>(P, WhT, out);
}
